// MyAttention_28930899706337
// MI455X (gfx1250) — compile-verified, matrix operand order checked by reading, not run on hardware
//
#include <hip/hip_runtime.h>
#include <hip/hip_bf16.h>

typedef __attribute__((ext_vector_type(16))) __bf16 v16bf;
typedef __attribute__((ext_vector_type(8)))  float  v8f;

#define B_DIM 2
#define C_DIM 512
#define S_DIM 256
#define E_DIM 512
#define M_TOT (B_DIM * C_DIM * S_DIM)          // 262144
#define NE    ((size_t)M_TOT * E_DIM)          // 134217728
#define NP    ((size_t)B_DIM * C_DIM * S_DIM * S_DIM)  // 67108864

union FragU { uint4 u4[2]; v16bf v; };
union Raw8 { uint4 u; __bf16 h[8]; };

// Fragment load from LDS tile (row stride in bf16 elements, must keep 16B align).
// Layout per CDNA5 ISA 7.12.2: lanes 0-15 hold k {0..7,16..23}, lanes 16-31 {8..15,24..31}.
__device__ __forceinline__ v16bf frag_ld(const __bf16* base, int stride, int r0, int lane) {
    const __bf16* p = base + (size_t)(r0 + (lane & 15)) * stride + ((lane & 16) ? 8 : 0);
    FragU f;
    f.u4[0] = *(const uint4*)(p);
    f.u4[1] = *(const uint4*)(p + 16);
    return f.v;
}

__device__ __forceinline__ v8f wmma_bf16(v16bf a, v16bf b, v8f c) {
    return __builtin_amdgcn_wmma_f32_16x16x32_bf16(false, a, false, b, (short)0, c, false, false);
}

__device__ __forceinline__ unsigned pack2(__bf16 a, __bf16 b) {
    return (unsigned)__builtin_bit_cast(unsigned short, a) |
          ((unsigned)__builtin_bit_cast(unsigned short, b) << 16);
}

// ---------------- Kernel 1: per-channel batch stats ----------------
__global__ __launch_bounds__(256) void mha_bn_stats(const float* __restrict__ x,
                                                    float* __restrict__ mean,
                                                    float* __restrict__ rstd) {
    const int c = blockIdx.x;
    const int SE = S_DIM * E_DIM;  // 131072
    float s = 0.f, s2 = 0.f;
    for (int b = 0; b < B_DIM; ++b) {
        const float4* p = (const float4*)(x + (size_t)(b * C_DIM + c) * SE);
        for (int i = threadIdx.x; i < SE / 4; i += 256) {
            float4 v = p[i];
            s  += v.x + v.y + v.z + v.w;
            s2 += v.x * v.x + v.y * v.y + v.z * v.z + v.w * v.w;
        }
    }
    __shared__ float rs[256], rq[256];
    rs[threadIdx.x] = s; rq[threadIdx.x] = s2;
    __syncthreads();
    for (int d = 128; d > 0; d >>= 1) {
        if ((int)threadIdx.x < d) { rs[threadIdx.x] += rs[threadIdx.x + d]; rq[threadIdx.x] += rq[threadIdx.x + d]; }
        __syncthreads();
    }
    if (threadIdx.x == 0) {
        const float n = (float)(B_DIM * SE);
        float m = rs[0] / n;
        float var = rq[0] / n - m * m;
        mean[c] = m;
        rstd[c] = rsqrtf(var + 1e-5f);
    }
}

// ---------------- Kernel 2: normalize + convert to bf16 ----------------
__global__ __launch_bounds__(256) void mha_bn_norm(const float* __restrict__ x,
                                                   const float* __restrict__ mean,
                                                   const float* __restrict__ rstd,
                                                   __bf16* __restrict__ xn) {
    size_t i4 = ((size_t)blockIdx.x * 256 + threadIdx.x) * 4;
    int c = (int)((i4 >> 17) & (C_DIM - 1));   // S*E = 2^17
    float m = mean[c], r = rstd[c];
    float4 v = *(const float4*)(x + i4);
    uint2 o;
    o.x = pack2((__bf16)((v.x - m) * r), (__bf16)((v.y - m) * r));
    o.y = pack2((__bf16)((v.z - m) * r), (__bf16)((v.w - m) * r));
    *(uint2*)(xn + i4) = o;
}

// ---------------- Kernel 3: fused q/k/v GEMM (bf16 WMMA) ----------------
// C[w] = xn(M=262144 x K=512) * W_w(512 x 512), outputs bf16
__global__ __launch_bounds__(256) void mha_qkv_gemm(const __bf16* __restrict__ xn,
                                                    const float* __restrict__ Qw,
                                                    const float* __restrict__ Kw,
                                                    const float* __restrict__ Vw,
                                                    __bf16* __restrict__ qo,
                                                    __bf16* __restrict__ ko,
                                                    __bf16* __restrict__ vo) {
    __shared__ __align__(16) __bf16 As[128 * 40];      // [m][kk], stride 40
    __shared__ __align__(16) __bf16 Ws[3][64 * 40];    // [n][kk], stride 40
    const int bn = blockIdx.x & 7;          // N split fastest -> L2 reuse of A panel
    const int bm = blockIdx.x >> 3;
    const int m0 = bm * 128, n0 = bn * 64;
    const int t = threadIdx.x, lane = t & 31, wv = t >> 5;
    const int wm = wv & 3, wn = wv >> 2;    // 4 m-waves x 2 n-waves
    v8f acc[3][4];
    #pragma unroll
    for (int w = 0; w < 3; ++w)
        #pragma unroll
        for (int i = 0; i < 4; ++i) acc[w][i] = (v8f){0.f,0.f,0.f,0.f,0.f,0.f,0.f,0.f};

    const int arow = t >> 2, aseg = t & 3;          // A-tile loader mapping
    const int wnn = t & 63, wkg = t >> 6;           // W-tile loader mapping

    for (int k0 = 0; k0 < E_DIM; k0 += 32) {
        // A rows 0..63 of the 128x32 tile, one b128 per thread
        *(uint4*)(&As[arow * 40 + aseg * 8]) =
            *(const uint4*)(xn + (size_t)(m0 + arow) * E_DIM + k0 + aseg * 8);
        // W tiles: f32 global [kk][n] -> bf16 LDS [n][kk] (transpose + convert)
        #pragma unroll
        for (int w = 0; w < 3; ++w) {
            const float* wp = (w == 0 ? Qw : w == 1 ? Kw : Vw)
                            + (size_t)(k0 + wkg * 8) * E_DIM + n0 + wnn;
            __bf16 tb[8];
            #pragma unroll
            for (int i = 0; i < 8; ++i) tb[i] = (__bf16)wp[(size_t)i * E_DIM];
            uint4 pk;
            pk.x = pack2(tb[0], tb[1]); pk.y = pack2(tb[2], tb[3]);
            pk.z = pack2(tb[4], tb[5]); pk.w = pack2(tb[6], tb[7]);
            *(uint4*)(&Ws[w][wnn * 40 + wkg * 8]) = pk;
        }
        __syncthreads();
        v16bf af0 = frag_ld(As, 40, wm * 32 + 0, lane);
        v16bf af1 = frag_ld(As, 40, wm * 32 + 16, lane);
        #pragma unroll
        for (int w = 0; w < 3; ++w) {
            v16bf bf0 = frag_ld(Ws[w], 40, wn * 32 + 0, lane);
            v16bf bf1 = frag_ld(Ws[w], 40, wn * 32 + 16, lane);
            acc[w][0] = wmma_bf16(af0, bf0, acc[w][0]);
            acc[w][1] = wmma_bf16(af0, bf1, acc[w][1]);
            acc[w][2] = wmma_bf16(af1, bf0, acc[w][2]);
            acc[w][3] = wmma_bf16(af1, bf1, acc[w][3]);
        }
        __syncthreads();
    }
    const int colb = n0 + wn * 32 + (lane & 15);
    const int rowb = m0 + wm * 32 + ((lane & 16) ? 8 : 0);
    #pragma unroll
    for (int w = 0; w < 3; ++w) {
        __bf16* op = (w == 0 ? qo : w == 1 ? ko : vo);
        #pragma unroll
        for (int i = 0; i < 2; ++i)
            #pragma unroll
            for (int j = 0; j < 2; ++j) {
                v8f a = acc[w][i * 2 + j];
                #pragma unroll
                for (int r = 0; r < 8; ++r)
                    op[(size_t)(rowb + i * 16 + r) * E_DIM + colb + j * 16] = (__bf16)a[r];
            }
    }
}

// ---------------- Kernel 4: batched scores = q*k^T / sqrt(E) ----------------
__global__ __launch_bounds__(256) void mha_scores_gemm(const __bf16* __restrict__ q,
                                                       const __bf16* __restrict__ k,
                                                       float* __restrict__ prod) {
    __shared__ __align__(16) __bf16 As[128 * 40];
    __shared__ __align__(16) __bf16 Bs[64 * 40];
    const int bc = blockIdx.y;                       // 1024 batches
    const int bm = blockIdx.x & 1, bn = blockIdx.x >> 1;   // 2 x 4 tiles
    const __bf16* A  = q + (size_t)bc * S_DIM * E_DIM;
    const __bf16* Bk = k + (size_t)bc * S_DIM * E_DIM;
    float* P = prod + (size_t)bc * S_DIM * S_DIM;
    const int m0 = bm * 128, n0 = bn * 64;
    const int t = threadIdx.x, lane = t & 31, wv = t >> 5;
    const int wm = wv & 3, wn = wv >> 2;
    v8f acc[4];
    #pragma unroll
    for (int i = 0; i < 4; ++i) acc[i] = (v8f){0.f,0.f,0.f,0.f,0.f,0.f,0.f,0.f};
    const int arow = t >> 2, aseg = t & 3;           // 64 rows x 4 segs
    const int brow = t >> 2, bseg = t & 3;           // 64 rows x 4 segs (threads 0..255)

    for (int k0 = 0; k0 < E_DIM; k0 += 32) {
        *(uint4*)(&As[arow * 40 + aseg * 8]) =
            *(const uint4*)(A + (size_t)(m0 + arow) * E_DIM + k0 + aseg * 8);
        if (brow < 64)
            *(uint4*)(&Bs[brow * 40 + bseg * 8]) =
                *(const uint4*)(Bk + (size_t)(n0 + brow) * E_DIM + k0 + bseg * 8);
        __syncthreads();
        v16bf af0 = frag_ld(As, 40, wm * 32 + 0, lane);
        v16bf af1 = frag_ld(As, 40, wm * 32 + 16, lane);
        v16bf bf0 = frag_ld(Bs, 40, wn * 32 + 0, lane);
        v16bf bf1 = frag_ld(Bs, 40, wn * 32 + 16, lane);
        acc[0] = wmma_bf16(af0, bf0, acc[0]);
        acc[1] = wmma_bf16(af0, bf1, acc[1]);
        acc[2] = wmma_bf16(af1, bf0, acc[2]);
        acc[3] = wmma_bf16(af1, bf1, acc[3]);
        __syncthreads();
    }
    const float scale = 0.044194173824159216f;  // 1/sqrt(512)
    const int colb = n0 + wn * 32 + (lane & 15);
    const int rowb = m0 + wm * 32 + ((lane & 16) ? 8 : 0);
    #pragma unroll
    for (int i = 0; i < 2; ++i)
        #pragma unroll
        for (int j = 0; j < 2; ++j) {
            v8f a = acc[i * 2 + j];
            #pragma unroll
            for (int r = 0; r < 8; ++r)
                P[(size_t)(rowb + i * 16 + r) * S_DIM + colb + j * 16] = a[r] * scale;
        }
}

// ---------------- Kernel 5: online softmax over channel axis ----------------
__global__ __launch_bounds__(256) void mha_softmax_c(const float* __restrict__ prod,
                                                     __bf16* __restrict__ sm) {
    size_t p = (size_t)blockIdx.x * 256 + threadIdx.x;   // over B*S*S = 131072
    int b  = (int)(p >> 16);
    int st = (int)(p & 65535);
    const size_t cs = (size_t)S_DIM * S_DIM;             // 65536 channel stride
    const float* base = prod + (size_t)b * C_DIM * cs + st;
    // single pass: running max + rescaled running sum
    float mx = -3.0e38f, sum = 0.f;
    for (int c = 0; c < C_DIM; ++c) {
        float v = base[(size_t)c * cs];
        float m2 = fmaxf(mx, v);
        sum = sum * __expf(mx - m2) + __expf(v - m2);
        mx = m2;
    }
    float inv = 1.0f / sum;
    __bf16* ob = sm + (size_t)b * C_DIM * cs + st;
    for (int c = 0; c < C_DIM; ++c)
        ob[(size_t)c * cs] = (__bf16)(__expf(base[(size_t)c * cs] - mx) * inv);
}

// ---------------- Kernel 6: attention = sm * v + xn ----------------
__global__ __launch_bounds__(256) void mha_attn_gemm(const __bf16* __restrict__ sm,
                                                     const __bf16* __restrict__ v,
                                                     const __bf16* __restrict__ xn,
                                                     float* __restrict__ out) {
    __shared__ __align__(16) __bf16 As[128 * 40];
    __shared__ __align__(16) __bf16 Bs[64 * 40];
    const int bc = blockIdx.y;                            // 1024
    const int bm = blockIdx.x & 1, bn = blockIdx.x >> 1;  // 2 x 8 tiles
    const __bf16* A  = sm + (size_t)bc * S_DIM * S_DIM;
    const __bf16* Bv = v  + (size_t)bc * S_DIM * E_DIM;
    const __bf16* Xr = xn + (size_t)bc * S_DIM * E_DIM;
    float* O = out + (size_t)bc * S_DIM * E_DIM;
    const int m0 = bm * 128, n0 = bn * 64;
    const int t = threadIdx.x, lane = t & 31, wv = t >> 5;
    const int wm = wv & 3, wn = wv >> 2;
    v8f acc[4];
    #pragma unroll
    for (int i = 0; i < 4; ++i) acc[i] = (v8f){0.f,0.f,0.f,0.f,0.f,0.f,0.f,0.f};
    const int arow = t >> 2, aseg = t & 3;
    const int vrow = t >> 3, vfg = t & 7;   // 32 k-rows x 8 f-groups

    for (int k0 = 0; k0 < S_DIM; k0 += 32) {              // K = S = 256
        *(uint4*)(&As[arow * 40 + aseg * 8]) =
            *(const uint4*)(A + (size_t)(m0 + arow) * S_DIM + k0 + aseg * 8);
        {   // v[t][f] -> LDS [n=f][kk=t]: coalesced b128 row read, LDS b16 scatter
            Raw8 r8;
            r8.u = *(const uint4*)(Bv + (size_t)(k0 + vrow) * E_DIM + n0 + vfg * 8);
            #pragma unroll
            for (int i = 0; i < 8; ++i)
                Bs[(vfg * 8 + i) * 40 + vrow] = r8.h[i];
        }
        __syncthreads();
        v16bf af0 = frag_ld(As, 40, wm * 32 + 0, lane);
        v16bf af1 = frag_ld(As, 40, wm * 32 + 16, lane);
        v16bf bf0 = frag_ld(Bs, 40, wn * 32 + 0, lane);
        v16bf bf1 = frag_ld(Bs, 40, wn * 32 + 16, lane);
        acc[0] = wmma_bf16(af0, bf0, acc[0]);
        acc[1] = wmma_bf16(af0, bf1, acc[1]);
        acc[2] = wmma_bf16(af1, bf0, acc[2]);
        acc[3] = wmma_bf16(af1, bf1, acc[3]);
        __syncthreads();
    }
    const int colb = n0 + wn * 32 + (lane & 15);
    const int rowb = m0 + wm * 32 + ((lane & 16) ? 8 : 0);
    #pragma unroll
    for (int i = 0; i < 2; ++i)
        #pragma unroll
        for (int j = 0; j < 2; ++j) {
            v8f a = acc[i * 2 + j];
            #pragma unroll
            for (int r = 0; r < 8; ++r) {
                size_t idx = (size_t)(rowb + i * 16 + r) * E_DIM + colb + j * 16;
                __builtin_nontemporal_store(a[r] + (float)Xr[idx], &O[idx]);
            }
        }
}

extern "C" void kernel_launch(void* const* d_in, const int* in_sizes, int n_in,
                              void* d_out, int out_size, void* d_ws, size_t ws_size,
                              hipStream_t stream) {
    const float* x  = (const float*)d_in[0];
    const float* Qw = (const float*)d_in[1];
    const float* Kw = (const float*)d_in[2];
    const float* Vw = (const float*)d_in[3];
    float* out = (float*)d_out;

    char* ws = (char*)d_ws;
    float*  mean = (float*)ws;            // 512 f32
    float*  rstd = mean + 512;            // 512 f32
    __bf16* xn   = (__bf16*)(ws + 4096);
    __bf16* qb   = xn + NE;
    __bf16* kb   = qb + NE;
    __bf16* vb   = kb + NE;
    float*  prod = (float*)(vb + NE);
    __bf16* smb  = (__bf16*)(prod + NP);

    mha_bn_stats<<<C_DIM, 256, 0, stream>>>(x, mean, rstd);
    mha_bn_norm<<<(unsigned)(NE / (256 * 4)), 256, 0, stream>>>(x, mean, rstd, xn);
    mha_qkv_gemm<<<dim3((M_TOT / 128) * 8), 256, 0, stream>>>(xn, Qw, Kw, Vw, qb, kb, vb);
    mha_scores_gemm<<<dim3(8, B_DIM * C_DIM), 256, 0, stream>>>(qb, kb, prod);
    mha_softmax_c<<<(B_DIM * S_DIM * S_DIM) / 256, 256, 0, stream>>>(prod, smb);
    mha_attn_gemm<<<dim3(16, B_DIM * C_DIM), 256, 0, stream>>>(smb, vb, xn, out);
}
